// MultiLayerLSTM_20117626814480
// MI455X (gfx1250) — matrix operand order repaired and checked by reading, not compiled, not run on hardware
//
#include <hip/hip_runtime.h>
#include <hip/hip_bf16.h>

#define LAYERS 2
#define BATCH  32
#define TSEQ   512
#define DIM    1024
#define HID    1024
#define KDIM   (HID + DIM)      // 2048
#define NGATE  (4 * HID)        // 4096
#define NSTRIP (NGATE / 16)     // 256 column strips
#define KBLK   (KDIM / 32)      // 64 k-blocks of 32
#define NWG    64
#define TPB    128              // 4 waves per WG -> 256 waves total
#define ROWB   4112             // padded LDS row stride in bytes (2048 bf16 + 16B pad)
#define SMEM_BYTES (BATCH * ROWB)

typedef __attribute__((ext_vector_type(16))) __bf16 v16bf;
typedef __attribute__((ext_vector_type(8)))  float  v8f;

union V16U { v16bf v; uint4 q[2]; };

__device__ __forceinline__ unsigned short f32_to_bf16(float f) {
  unsigned u = __float_as_uint(f);
  u += 0x7FFFu + ((u >> 16) & 1u);   // round-to-nearest-even
  return (unsigned short)(u >> 16);
}

__device__ __forceinline__ float sigmoidf_(float x) {
  return 1.0f / (1.0f + __expf(-x));
}
__device__ __forceinline__ float tanhf_(float x) {
  // tanh(x) = 2*sigmoid(2x) - 1, stays on v_exp_f32 (no ocml call)
  return 2.0f / (1.0f + __expf(-2.0f * x)) - 1.0f;
}

// ---------------- weight packing: f32 (L,K,4H) -> bf16 WMMA-B tile layout ----
// tile index = (l*NSTRIP + cs)*KBLK + kb ; within tile: lane = (n&15) + 16*((k>>4)&1),
// element e = k&15  (contiguous k order, k = 16*(lane>>4) + e; A tiles from LDS use the same)
__global__ void pack_w_kernel(const float* __restrict__ W, unsigned short* __restrict__ wp) {
  long idx = (long)blockIdx.x * blockDim.x + threadIdx.x;
  const long total = (long)LAYERS * KDIM * NGATE;
  if (idx >= total) return;
  int n = (int)(idx % NGATE);
  int k = (int)((idx / NGATE) % KDIM);
  int l = (int)(idx / ((long)NGATE * KDIM));
  int cs = n >> 4, kb = k >> 5;
  int lane = (n & 15) + (((k >> 4) & 1) << 4);
  int e = k & 15;
  long tile = ((long)(l * NSTRIP + cs)) * KBLK + kb;
  wp[tile * 512 + lane * 16 + e] = f32_to_bf16(W[idx]);
}

__global__ void pack_x_kernel(const float* __restrict__ x, unsigned short* __restrict__ xp, long n) {
  long idx = (long)blockIdx.x * blockDim.x + threadIdx.x;
  if (idx < n) xp[idx] = f32_to_bf16(x[idx]);
}

// ---------------- device-wide generation barrier -----------------------------
__device__ __forceinline__ void grid_barrier(unsigned* bar, unsigned gen) {
  __syncthreads();
  if (threadIdx.x == 0) {
    __threadfence();
    atomicAdd(bar, 1u);
    const unsigned target = gen * NWG;
    while (__hip_atomic_load(bar, __ATOMIC_ACQUIRE, __HIP_MEMORY_SCOPE_AGENT) < target) {
      __builtin_amdgcn_s_sleep(2);
    }
    __threadfence();
  }
  __syncthreads();
}

// ---------------- LDS helpers ------------------------------------------------
__device__ __forceinline__ v16bf load_a_tile(const unsigned char* base) {
  // A-matrix 16x32 bf16 per-lane layout: lane holds 16 contiguous k elements
  V16U u;
  u.q[0] = *(const uint4*)(base);
  u.q[1] = *(const uint4*)(base + 16);
  return u.v;
}

__device__ __forceinline__ v16bf load_b_tile(const unsigned short* p) {
  V16U u;
  u.q[0] = *(const uint4*)(p);
  u.q[1] = *(const uint4*)(p + 8);
  return u.v;
}

// ---------------- persistent LSTM kernel -------------------------------------
__global__ __launch_bounds__(TPB) void lstm_kernel(
    const unsigned short* __restrict__ wp,    // packed bf16 weights
    const unsigned short* __restrict__ xp,    // bf16 x, (B,T,D) layout
    const float* __restrict__ bias,           // (L, 4H)
    unsigned short* __restrict__ hbf,         // bf16 h state, (L,B,H)
    float* __restrict__ cst,                  // f32 c state,  (L,B,H)
    float* __restrict__ gates,                // f32 scratch,  (B,4H)
    unsigned* __restrict__ bar,
    float* __restrict__ out)                  // outputs (B,T,H) | h_n (L,B,H) | c_n (L,B,H)
{
  extern __shared__ __align__(64) unsigned char smem[];
  const int tid  = threadIdx.x;
  const int lane = tid & 31;
  const int wave = blockIdx.x * (TPB / 32) + (tid >> 5);   // 0..255 == column strip
  const unsigned lds_base = (unsigned)(size_t)(&smem[0]);
  unsigned gen = 0;

  const int c    = lane >> 4;        // half-wave selector
  const int mrow = lane & 15;
  const int ncol = lane & 15;

  for (int t = 0; t < TSEQ; ++t) {
    for (int l = 0; l < LAYERS; ++l) {
      // ---- stage comb = [h_prev(l), input] (32 x 2048 bf16) into LDS, async path ----
      const unsigned short* hsrc = hbf + (size_t)l * BATCH * HID;
      for (int ch = tid; ch < BATCH * 256; ch += TPB) {     // 16B chunks
        const int b  = ch >> 8;
        const int c0 = ch & 255;
        const void* src;
        if (c0 < 128)      src = hsrc + ((size_t)b * HID + c0 * 8);
        else if (l == 0)   src = xp + (((size_t)b * TSEQ + t) * DIM + (c0 - 128) * 8);
        else               src = hbf + ((size_t)b * HID + (c0 - 128) * 8);   // layer-0 h at this t
        unsigned dst = lds_base + (unsigned)(b * ROWB + c0 * 16);
        asm volatile("global_load_async_to_lds_b128 %0, %1, off"
                     :: "v"(dst), "v"(src) : "memory");
      }
      asm volatile("s_wait_asynccnt 0" ::: "memory");
      __syncthreads();

      // ---- GEMM: strip = wave, acc over K, two M-tiles (batch 0-15, 16-31) ----
      const unsigned short* bp = wp + ((size_t)(l * NSTRIP + wave) * KBLK) * 512 + lane * 16;
      const unsigned char* a0base = smem + mrow * ROWB + c * 32;
      const unsigned char* a1base = a0base + 16 * ROWB;
      v8f acc0 = {}, acc1 = {};
#pragma unroll 8
      for (int kb = 0; kb < KBLK; ++kb) {
        v16bf bm = load_b_tile(bp);  bp += 512;
        v16bf a0 = load_a_tile(a0base + kb * 64);
        v16bf a1 = load_a_tile(a1base + kb * 64);
        acc0 = __builtin_amdgcn_wmma_f32_16x16x32_bf16(false, a0, false, bm,
                                                       (short)0, acc0, false, false);
        acc1 = __builtin_amdgcn_wmma_f32_16x16x32_bf16(false, a1, false, bm,
                                                       (short)0, acc1, false, false);
      }

      // ---- write gates (+bias); C/D layout: vgpr r -> M = r + 8*(lane>=16) ----
      const int col = wave * 16 + ncol;
      const float bia = bias[l * NGATE + col];
      const int rbase = c * 8;
#pragma unroll
      for (int r = 0; r < 8; ++r) {
        gates[(size_t)(rbase + r)      * NGATE + col] = acc0[r] + bia;
        gates[(size_t)(rbase + r + 16) * NGATE + col] = acc1[r] + bia;
      }

      grid_barrier(bar, ++gen);

      // ---- elementwise cell update, spread over all 256 waves ----
      // wave -> unit strip (wave>>2), 8 batch rows ((wave&3)*8 .. +7)
      {
        const int j = (wave >> 2) * 16 + ncol;
        const int rowbase = (wave & 3) * 8;
        float*          crow = cst + (size_t)l * BATCH * HID;
        unsigned short* hrow = hbf + (size_t)l * BATCH * HID;
#pragma unroll
        for (int it = 0; it < 4; ++it) {
          const int b = rowbase + c + it * 2;
          const float* gr = gates + (size_t)b * NGATE;
          const float f = sigmoidf_(gr[j]);
          const float i = sigmoidf_(gr[HID + j]);
          const float g = tanhf_(gr[2 * HID + j]);
          const float o = sigmoidf_(gr[3 * HID + j]);
          const size_t ci = (size_t)b * HID + j;
          const float cn = f * crow[ci] + i * g;
          const float hn = o * tanhf_(cn);
          crow[ci] = cn;
          hrow[ci] = f32_to_bf16(hn);
          if (l == LAYERS - 1) out[((size_t)b * TSEQ + t) * HID + j] = hn;
          if (t == TSEQ - 1) {
            const size_t o2 = (size_t)BATCH * TSEQ * HID;
            out[o2 + ((size_t)l * BATCH + b) * HID + j] = hn;
            out[o2 + (size_t)LAYERS * BATCH * HID + ((size_t)l * BATCH + b) * HID + j] = cn;
          }
        }
      }

      grid_barrier(bar, ++gen);
    }
  }
}

// ---------------- host launch ------------------------------------------------
extern "C" void kernel_launch(void* const* d_in, const int* in_sizes, int n_in,
                              void* d_out, int out_size, void* d_ws, size_t ws_size,
                              hipStream_t stream) {
  (void)in_sizes; (void)n_in; (void)out_size; (void)ws_size;
  const float* x = (const float*)d_in[0];
  const float* W = (const float*)d_in[1];
  const float* b = (const float*)d_in[2];
  float* out = (float*)d_out;

  unsigned char* ws = (unsigned char*)d_ws;
  const size_t WPB = (size_t)LAYERS * KDIM * NGATE * 2;   // 32 MB packed weights
  const size_t XPB = (size_t)BATCH * TSEQ * DIM * 2;      // 32 MB packed x
  const size_t HB  = (size_t)LAYERS * BATCH * HID * 2;    // 128 KB h (bf16)
  const size_t CB  = (size_t)LAYERS * BATCH * HID * 4;    // 256 KB c (f32)
  const size_t GB  = (size_t)BATCH * NGATE * 4;           // 512 KB gates

  unsigned short* wp    = (unsigned short*)(ws);
  unsigned short* xp    = (unsigned short*)(ws + WPB);
  unsigned short* hbf   = (unsigned short*)(ws + WPB + XPB);
  float*          cst   = (float*)(ws + WPB + XPB + HB);
  float*          gates = (float*)(ws + WPB + XPB + HB + CB);
  unsigned*       bar   = (unsigned*)(ws + WPB + XPB + HB + CB + GB);

  // deterministic state reset each call (graph-capturable)
  hipMemsetAsync(hbf, 0, HB, stream);
  hipMemsetAsync(cst, 0, CB, stream);
  hipMemsetAsync(bar, 0, sizeof(unsigned), stream);

  const long wtotal = (long)LAYERS * KDIM * NGATE;
  const long xtotal = (long)BATCH * TSEQ * DIM;
  pack_w_kernel<<<(int)((wtotal + 255) / 256), 256, 0, stream>>>(W, wp);
  pack_x_kernel<<<(int)((xtotal + 255) / 256), 256, 0, stream>>>(x, xp, xtotal);

  lstm_kernel<<<NWG, TPB, SMEM_BYTES, stream>>>(wp, xp, b, hbf, cst, gates, bar, out);
}
